// StructureAwareGraphAttention_41016937677102
// MI455X (gfx1250) — matrix operand order repaired and checked by reading, not compiled, not run on hardware
//
#include <hip/hip_runtime.h>
#include <hip/hip_bf16.h>

// Problem constants (from reference)
#define CC 4
#define SS 4096
#define HH 128
#define NHEAD 4
#define HD 32
// 32^-0.5 * log2(e): fold softmax base-2 conversion into the Q projection
#define QSCALE 0.25501816522031243f
#define EPSW 1e-8f

typedef __attribute__((ext_vector_type(16))) __bf16 v16bf;
typedef __attribute__((ext_vector_type(8)))  float  v8f;
typedef __attribute__((ext_vector_type(2)))  __bf16 v2bf;

#if __has_builtin(__builtin_amdgcn_cvt_pk_bf16_f32)
#define HAVE_PKBF 1
#else
#define HAVE_PKBF 0
#endif

__device__ __forceinline__ v2bf f2bf2(float x, float y) {
#if HAVE_PKBF
  return __builtin_amdgcn_cvt_pk_bf16_f32(x, y);   // v_cvt_pk_bf16_f32
#else
  unsigned ux = __float_as_uint(x);
  ux += 0x7FFFu + ((ux >> 16) & 1u);
  unsigned uy = __float_as_uint(y);
  uy += 0x7FFFu + ((uy >> 16) & 1u);
  unsigned p = (ux >> 16) | (uy & 0xFFFF0000u);
  v2bf r; __builtin_memcpy(&r, &p, 4);
  return r;
#endif
}

__device__ __forceinline__ __bf16 f2bf(float x) {
  v2bf r = f2bf2(x, x);
  return r[0];
}

// 16 consecutive bf16 (32 B) -> v16bf
__device__ __forceinline__ v16bf ld16bf(const __bf16* p) {
  uint4 lo = *(const uint4*)p;
  uint4 hi = *(const uint4*)(p + 8);
  struct { uint4 a, b; } s{lo, hi};
  v16bf v;
  __builtin_memcpy(&v, &s, 32);
  return v;
}

// two 16-B chunks (8 bf16 each) -> v16bf  (A-operand pattern)
__device__ __forceinline__ v16bf ld2x8bf(const __bf16* c0, const __bf16* c1) {
  uint4 lo = *(const uint4*)c0;
  uint4 hi = *(const uint4*)c1;
  struct { uint4 a, b; } s{lo, hi};
  v16bf v;
  __builtin_memcpy(&v, &s, 32);
  return v;
}

__device__ __forceinline__ void ld_f4(float* dst, const float* p) {
  float4 v = *(const float4*)p;
  dst[0] = v.x; dst[1] = v.y; dst[2] = v.z; dst[3] = v.w;
}

// ---------------------------------------------------------------------------
// Kernel 0: one-shot f32 -> bf16 conversion of Wq|Wk|Wv|Wo (4 x 128x128) AND
// the activation x (4x4096x128).  All GEMM operands afterwards are contiguous
// bf16 global_load_b128 -- no per-tile conversion VALU work anywhere.
// ---------------------------------------------------------------------------
__global__ __launch_bounds__(256) void prep_kernel(
    const float* __restrict__ Wq, const float* __restrict__ Wk,
    const float* __restrict__ Wv, const float* __restrict__ Wo,
    const float* __restrict__ x,
    __bf16* __restrict__ Wb, __bf16* __restrict__ xb)
{
  const int idx = blockIdx.x * 256 + threadIdx.x;      // pair index
  if (idx < 32768) {                                   // 4 weight matrices
    const int i2 = idx * 2;
    const float* src = (i2 < 16384) ? Wq : (i2 < 32768) ? Wk
                     : (i2 < 49152) ? Wv : Wo;
    const int off = i2 & 16383;
    *(v2bf*)(Wb + i2) = f2bf2(src[off], src[off + 1]);
  } else {                                             // x: 2,097,152 elems
    const size_t j2 = (size_t)(idx - 32768) * 2;
    *(v2bf*)(xb + j2) = f2bf2(x[j2], x[j2 + 1]);
  }
}

// ---------------------------------------------------------------------------
// Kernel 1: fused QKV projection.  out = x @ W^T + b, per-head split to bf16.
//   Q,K stored [c][h][s][hd] row-major; V stored TRANSPOSED [c][h][hd][s].
//   Q is additionally scaled by 32^-0.5 * log2(e) (base-2 softmax).
// One wave computes two 16x16 output tiles; 4 waves/block cover 8 feature tiles.
// ---------------------------------------------------------------------------
__global__ __launch_bounds__(128) void proj_qkv_kernel(
    const __bf16* __restrict__ xb, const __bf16* __restrict__ Wb,
    const float* __restrict__ bq, const float* __restrict__ bk,
    const float* __restrict__ bv,
    __bf16* __restrict__ Qb, __bf16* __restrict__ Kb, __bf16* __restrict__ Vt)
{
  const int lane = threadIdx.x & 31;
  const int wave = threadIdx.x >> 5;
  const int lm = lane & 15;          // A-row / B-col / D-col
  const int kh = lane >> 4;          // lane half
  const int g0 = blockIdx.x * 16;    // global row base (c*S + s flattened)
  const int mat = blockIdx.y;        // 0=Q 1=K 2=V
  const __bf16* W   = Wb + (size_t)mat * HH * HH;
  const float* bias = (mat == 0) ? bq : (mat == 1) ? bk : bv;

  const int oc[2] = { wave * 32, wave * 32 + 16 };
  v8f acc[2] = { v8f{}, v8f{} };

#pragma unroll
  for (int k0 = 0; k0 < HH; k0 += 32) {
    // A operand: row lm, 16 contiguous K at k0+kh*16 (bf16 direct)
    const __bf16* xr = xb + (size_t)(g0 + lm) * HH + k0 + kh * 16;
    v16bf a = ld16bf(xr);
#pragma unroll
    for (int t = 0; t < 2; ++t) {
      // B[k][n] = W[n][k]; lane col n = oc+lm reads 16 contiguous k (bf16)
      v16bf b = ld16bf(W + (size_t)(oc[t] + lm) * HH + k0 + kh * 16);
      acc[t] = __builtin_amdgcn_wmma_f32_16x16x32_bf16(
          false, a, false, b, (short)0, acc[t], false, false);
    }
  }

#pragma unroll
  for (int t = 0; t < 2; ++t) {
    const int o  = oc[t] + lm;        // feature column (D-layout col = lm)
    const float bi = bias[o];
    const int hI = o >> 5, hdI = o & 31;
#pragma unroll
    for (int r = 0; r < 8; ++r) {
      const int g = g0 + r + 8 * kh;  // D-layout row
      const int c = g >> 12;          // / 4096
      const int s = g & (SS - 1);
      float val = acc[t][r] + bi;
      if (mat == 0) val *= QSCALE;
      if (mat == 2)
        Vt[(((size_t)c * NHEAD + hI) * HD + hdI) * SS + s] = f2bf(val);
      else if (mat == 1)
        Kb[(((size_t)c * NHEAD + hI) * SS + s) * HD + hdI] = f2bf(val);
      else
        Qb[(((size_t)c * NHEAD + hI) * SS + s) * HD + hdI] = f2bf(val);
    }
  }
}

// ---------------------------------------------------------------------------
// Kernel 2: flash attention, base-2 softmax, additive log2(edge_weight) bias.
// Block = 256 threads = 8 waves: wave w -> head (w&3), q-subtile (w>>2).
// Block covers 32 q rows; edge-bias tile (32x32) is streamed from HBM exactly
// once, logged, and staged TRANSPOSED+padded in LDS shared by all 4 heads
// (4x HBM traffic cut; bias reads become ds_load_b128).
// Edge loads are register-pipelined one step ahead to hide HBM latency.
// Softmax row-sum is computed on the matrix pipe: P @ ones via one extra WMMA
// (no cross-lane shuffles); only the row-max keeps a 4-step butterfly.
// ---------------------------------------------------------------------------
__global__ __launch_bounds__(256) void flash_kernel(
    const __bf16* __restrict__ Qb, const __bf16* __restrict__ Kb,
    const __bf16* __restrict__ Vt, const float* __restrict__ ew,
    __bf16* __restrict__ Ob)
{
  __shared__ __align__(16) float  BiasT[32][36];     // [col][row], padded
  __shared__ __align__(16) __bf16 Pbuf[8][16][32];   // per-wave P tile

  const int tid  = threadIdx.x;
  const int lane = tid & 31, wave = tid >> 5;
  const int h  = wave & 3, qt = wave >> 2;
  const int lm = lane & 15, kh = lane >> 4;
  const int c  = blockIdx.y;
  const int qb = blockIdx.x * 32;       // block q base
  const int q0 = qb + qt * 16;          // wave q base

  const __bf16* Qp = Qb + ((size_t)c * NHEAD + h) * SS * HD;
  const __bf16* Kp = Kb + ((size_t)c * NHEAD + h) * SS * HD;
  const __bf16* Vp = Vt + ((size_t)c * NHEAD + h) * HD * SS; // [hd][s]
  const float*  Ep = ew + (size_t)c * SS * SS;

  // Q A-operand: row = lm, 16 contiguous K at kh*16 (pre-scaled)
  const __bf16* qr = Qp + (size_t)(q0 + lm) * HD + kh * 16;
  v16bf qa = ld16bf(qr);

  // constant all-ones B operand (bf16 1.0 packed) for row-sum WMMA
  v16bf onesb;
  {
    unsigned u[8];
#pragma unroll
    for (int i = 0; i < 8; ++i) u[i] = 0x3F803F80u;
    __builtin_memcpy(&onesb, u, 32);
  }

  v8f o0 = {}, o1 = {};
  float m[8], l[8];
#pragma unroll
  for (int r = 0; r < 8; ++r) { m[r] = -3.0e38f; l[r] = 0.f; }

  // pipeline: preload edge tile for j0 = 0 (4 floats/thread, row-coalesced)
  float ewreg[4];
#pragma unroll
  for (int i = 0; i < 4; ++i)
    ewreg[i] = Ep[(size_t)(qb + wave + i * 8) * SS + lane];

  for (int j0 = 0; j0 < SS; j0 += 32) {
    __syncthreads();                    // previous-step bias fully consumed
    // transform + stage transposed: BiasT[col][row] = log2(max(ew,eps))
#pragma unroll
    for (int i = 0; i < 4; ++i)
      BiasT[lane][wave + i * 8] = __builtin_amdgcn_logf(fmaxf(ewreg[i], EPSW));
    __syncthreads();
    if (j0 + 32 < SS) {                 // issue next tile's loads now; the
#pragma unroll                          //   wait lands at next loop top
      for (int i = 0; i < 4; ++i)
        ewreg[i] = Ep[(size_t)(qb + wave + i * 8) * SS + (j0 + 32) + lane];
    }

    // scores: S[mq][nk] = sum_d Q[mq][d] * K[j0+nk][d]   (base-2 scaled)
    v16bf kb0 = ld16bf(Kp + (size_t)(j0 + lm)      * HD + kh * 16);
    v16bf kb1 = ld16bf(Kp + (size_t)(j0 + 16 + lm) * HD + kh * 16);
    // V B-operands issued early so their latency hides under the softmax
    v16bf vb0 = ld16bf(Vp + (size_t)lm        * SS + j0 + kh * 16);
    v16bf vb1 = ld16bf(Vp + (size_t)(16 + lm) * SS + j0 + kh * 16);
    v8f s0 = {}, s1 = {};
    s0 = __builtin_amdgcn_wmma_f32_16x16x32_bf16(false, qa, false, kb0,
                                                 (short)0, s0, false, false);
    s1 = __builtin_amdgcn_wmma_f32_16x16x32_bf16(false, qa, false, kb1,
                                                 (short)0, s1, false, false);

    // vectorized bias fetch: 8 consecutive rows per column -> ds_load_b128 x4
    float bA[8], bB[8];
    {
      const float* bp0 = &BiasT[lm][qt * 16 + 8 * kh];
      const float* bp1 = &BiasT[16 + lm][qt * 16 + 8 * kh];
      ld_f4(bA, bp0); ld_f4(bA + 4, bp0 + 4);
      ld_f4(bB, bp1); ld_f4(bB + 4, bp1 + 4);
    }

    float al[8];
#pragma unroll
    for (int r = 0; r < 8; ++r) {
      const int Ml = r + 8 * kh;        // row within 16-row wave tile
      const float a0 = s0[r] + bA[r];
      const float a1 = s1[r] + bB[r];
      // row max over the 16 lanes of this half (xor 1/2/4/8 stays in half)
      float mx = fmaxf(a0, a1);
      mx = fmaxf(mx, __shfl_xor(mx, 1, 32));
      mx = fmaxf(mx, __shfl_xor(mx, 2, 32));
      mx = fmaxf(mx, __shfl_xor(mx, 4, 32));
      mx = fmaxf(mx, __shfl_xor(mx, 8, 32));
      const float mn = fmaxf(m[r], mx);
      const float p0 = __builtin_amdgcn_exp2f(a0 - mn);
      const float p1 = __builtin_amdgcn_exp2f(a1 - mn);
      al[r] = __builtin_amdgcn_exp2f(m[r] - mn);
      m[r] = mn;
      o0[r] *= al[r];
      o1[r] *= al[r];
      // P (D-layout) -> LDS row-major for A-layout reload
      v2bf pk = f2bf2(p0, p1);
      Pbuf[wave][Ml][lm]      = pk[0];
      Pbuf[wave][Ml][16 + lm] = pk[1];
    }
    asm volatile("s_wait_dscnt 0" ::: "memory");
    __builtin_amdgcn_wave_barrier();

    const __bf16* pr = &Pbuf[wave][lm][kh * 16];
    v16bf pa = ld16bf(pr);
    // row-sum on the matrix pipe: ls = P @ ones  (every lane gets its rows)
    v8f ls = {};
    ls = __builtin_amdgcn_wmma_f32_16x16x32_bf16(false, pa, false, onesb,
                                                 (short)0, ls, false, false);
    // O += P @ V: B[k][n] = V[j0+k][n] = Vt[n][j0+k] (contiguous in k)
    o0 = __builtin_amdgcn_wmma_f32_16x16x32_bf16(false, pa, false, vb0,
                                                 (short)0, o0, false, false);
    o1 = __builtin_amdgcn_wmma_f32_16x16x32_bf16(false, pa, false, vb1,
                                                 (short)0, o1, false, false);
#pragma unroll
    for (int r = 0; r < 8; ++r)
      l[r] = l[r] * al[r] + ls[r];
  }

#pragma unroll
  for (int r = 0; r < 8; ++r) {
    const float inv = 1.0f / l[r];
    const int Ml = r + 8 * kh;
    const size_t base = ((size_t)c * SS + (q0 + Ml)) * HH + h * HD;
    v2bf pk = f2bf2(o0[r] * inv, o1[r] * inv);
    Ob[base + lm]      = pk[0];
    Ob[base + 16 + lm] = pk[1];
  }
}

// ---------------------------------------------------------------------------
// Kernel 3: output projection  out = Ob @ Wo^T + bo  (f32 result)
// ---------------------------------------------------------------------------
__global__ __launch_bounds__(128) void out_proj_kernel(
    const __bf16* __restrict__ Ob, const __bf16* __restrict__ Wob,
    const float* __restrict__ bo, float* __restrict__ out)
{
  const int lane = threadIdx.x & 31, wave = threadIdx.x >> 5;
  const int lm = lane & 15, kh = lane >> 4;
  const int g0 = blockIdx.x * 16;
  const int oc[2] = { wave * 32, wave * 32 + 16 };
  v8f acc[2] = { v8f{}, v8f{} };

#pragma unroll
  for (int k0 = 0; k0 < HH; k0 += 32) {
    const __bf16* ar = Ob + (size_t)(g0 + lm) * HH + k0 + kh * 16;
    v16bf a = ld16bf(ar);
#pragma unroll
    for (int t = 0; t < 2; ++t) {
      v16bf b = ld16bf(Wob + (size_t)(oc[t] + lm) * HH + k0 + kh * 16);
      acc[t] = __builtin_amdgcn_wmma_f32_16x16x32_bf16(
          false, a, false, b, (short)0, acc[t], false, false);
    }
  }
#pragma unroll
  for (int t = 0; t < 2; ++t) {
    const float bi = bo[oc[t] + lm];
#pragma unroll
    for (int r = 0; r < 8; ++r) {
      const int g = g0 + r + 8 * kh;
      out[(size_t)g * HH + oc[t] + lm] = acc[t][r] + bi;
    }
  }
}

// ---------------------------------------------------------------------------
extern "C" void kernel_launch(void* const* d_in, const int* in_sizes, int n_in,
                              void* d_out, int out_size, void* d_ws, size_t ws_size,
                              hipStream_t stream)
{
  const float* x  = (const float*)d_in[0];
  const float* ew = (const float*)d_in[1];
  const float* Wq = (const float*)d_in[2];
  const float* bq = (const float*)d_in[3];
  const float* Wk = (const float*)d_in[4];
  const float* bk = (const float*)d_in[5];
  const float* Wv = (const float*)d_in[6];
  const float* bv = (const float*)d_in[7];
  const float* Wo = (const float*)d_in[8];
  const float* bo = (const float*)d_in[9];
  float* out = (float*)d_out;

  char* ws = (char*)d_ws;
  const size_t sz = (size_t)CC * NHEAD * SS * HD * 2;  // 4 MB each
  __bf16* Qb = (__bf16*)(ws);
  __bf16* Kb = (__bf16*)(ws + sz);
  __bf16* Vt = (__bf16*)(ws + 2 * sz);
  __bf16* Ob = (__bf16*)(ws + 3 * sz);
  __bf16* Wb = (__bf16*)(ws + 4 * sz);                 // 4 x 32 KB bf16 weights
  __bf16* xb = (__bf16*)(ws + 4 * sz + 4 * HH * HH * 2); // 4 MB bf16 x

  // pairs: 32768 (weights) + 1,048,576 (x) = 1,081,344 -> 4224 blocks
  prep_kernel<<<4224, 256, 0, stream>>>(Wq, Wk, Wv, Wo, x, Wb, xb);

  dim3 gp(CC * SS / 16, 3);
  proj_qkv_kernel<<<gp, 128, 0, stream>>>(xb, Wb, bq, bk, bv, Qb, Kb, Vt);

  dim3 gf(SS / 32, CC);
  flash_kernel<<<gf, 256, 0, stream>>>(Qb, Kb, Vt, ew, Ob);

  out_proj_kernel<<<CC * SS / 16, 128, 0, stream>>>(Ob, Wb + 3 * HH * HH, bo, out);
}
